// GlobalReconstructor_51539607552758
// MI455X (gfx1250) — matrix operand order repaired and checked by reading, not compiled, not run on hardware
//
#include <hip/hip_runtime.h>
#include <hip/hip_bf16.h>

// Problem constants: T=48, B=512, D=1024, H=1024, steps = T-1 = 47, M = 47*512 = 24064
typedef __attribute__((ext_vector_type(16))) __bf16 v16bf;
typedef __attribute__((ext_vector_type(8)))  float  v8f;
typedef __attribute__((ext_vector_type(4)))  unsigned int v4u;
typedef __attribute__((ext_vector_type(8)))  int v8i;
typedef __attribute__((ext_vector_type(4)))  int v4i;

union FragAB { v16bf v; unsigned int u[8]; };
union FragC  { v8f   v; float        f[8]; };

__device__ __forceinline__ unsigned short f32_to_bf16_rne(float x) {
  unsigned int u = __float_as_uint(x);
  u += 0x7FFFu + ((u >> 16) & 1u);
  return (unsigned short)(u >> 16);
}

// ---- TDM: async 2D bf16 tile load (rows x cols), global -> LDS -------------
// D# packing per CDNA5 ISA (2D tensor, groups 2/3 zero).
// This toolchain exposes the 6-arg builtin form:
//   (uint32x4 g0, int32x8 g1, int32x4 g2, int32x4 g3, int32x8 extra, i32 cpol)
__device__ __forceinline__ void tdm_load_2d_bf16(unsigned int lds_off_bytes,
                                                 const unsigned short* gptr,
                                                 int tile_rows, int tile_cols,
                                                 int tensor_rows, int tensor_cols,
                                                 int row_stride_elems) {
  const unsigned long long ga = (unsigned long long)(const void*)gptr;
  v4u g0;
  g0[0] = 1u;                                      // count=1, user descriptor
  g0[1] = lds_off_bytes;                           // lds_addr (bytes)
  g0[2] = (unsigned int)ga;                        // global_addr[31:0]
  g0[3] = (unsigned int)((ga >> 32) & 0x01FFFFFFull) | 0x80000000u; // [56:32] | type=2
  v8i g1;
  g1[0] = (int)(1u << 16);                         // wg_mask=0, data_size=1 (2 bytes)
  g1[1] = (int)((unsigned int)tensor_cols << 16);  // tensor_dim0[15:0] (atomic_addr=0)
  g1[2] = (int)(((unsigned int)tensor_cols >> 16) |
                ((unsigned int)tensor_rows << 16));             // dim0[31:16], dim1[15:0]
  g1[3] = (int)(((unsigned int)tensor_rows >> 16) |
                ((unsigned int)tile_cols << 16));               // dim1[31:16], tile_dim0
  g1[4] = tile_rows;                               // tile_dim1 (tile_dim2 = 0)
  g1[5] = row_stride_elems;                        // tensor_dim0_stride[31:0]
  g1[6] = 0;                                       // stride[47:32], dim1_stride[15:0]
  g1[7] = 0;
  const v4i z4 = {0, 0, 0, 0};
  const v8i z8 = {0, 0, 0, 0, 0, 0, 0, 0};
  __builtin_amdgcn_tensor_load_to_lds(g0, g1, z4, z4, z8, 0);
}

// ---------------------------------------------------------------- converts --
__global__ __launch_bounds__(256) void cvt_bf16_kernel(const float* __restrict__ in,
                                                       unsigned short* __restrict__ out,
                                                       int n) {
  int i = blockIdx.x * 256 + threadIdx.x;
  if (i < n) out[i] = f32_to_bf16_rne(in[i]);
}

// ------------------------------------------------------------------- pooled --
__global__ __launch_bounds__(256) void pool_kernel(const float* __restrict__ dh,
                                                   const int* __restrict__ captions,
                                                   unsigned short* __restrict__ pooled) {
  const int b = blockIdx.x;      // 512
  const int d = threadIdx.x;     // 256 threads, 4 d each
  float sum0 = 0.f, sum1 = 0.f, sum2 = 0.f, sum3 = 0.f, len = 0.f;
  for (int t = 0; t < 48; ++t) {
    const int tok = captions[t * 512 + b];
    const float m = (tok != 0 && tok != 2) ? 1.0f : 0.0f;
    len += m;
    const float* row = dh + ((size_t)t * 512 + b) * 1024;
    sum0 += m * row[d];
    sum1 += m * row[d + 256];
    sum2 += m * row[d + 512];
    sum3 += m * row[d + 768];
  }
  const float inv = 1.0f / len;
  unsigned short* out = pooled + (size_t)b * 1024;
  out[d]       = f32_to_bf16_rne(sum0 * inv);
  out[d + 256] = f32_to_bf16_rne(sum1 * inv);
  out[d + 512] = f32_to_bf16_rne(sum2 * inv);
  out[d + 768] = f32_to_bf16_rne(sum3 * inv);
}

// ------------------------------------------------------- materialize X bf16 --
__global__ __launch_bounds__(256) void xsprep_kernel(const float* __restrict__ dh,
                                                     const unsigned short* __restrict__ pooled,
                                                     unsigned short* __restrict__ X) {
  const int m = blockIdx.x;             // 24064
  const int t = m >> 9, b = m & 511;
  unsigned short* xrow = X + (size_t)m * 2048;
  const float* drow = dh + ((size_t)(t + 1) * 512 + b) * 1024;
  const unsigned short* prow = pooled + (size_t)b * 1024;
#pragma unroll
  for (int q = 0; q < 4; ++q) {
    const int k = threadIdx.x + q * 256;
    xrow[k] = f32_to_bf16_rne(drow[k]);
    xrow[1024 + k] = prow[k];
  }
}

// --------------------------------------------------------------------- init --
__global__ __launch_bounds__(256) void init_kernel(unsigned short* __restrict__ h0,
                                                   float* __restrict__ c0,
                                                   float* __restrict__ feats) {
  const int i = blockIdx.x * 256 + threadIdx.x;   // 512*1024 total
  h0[i] = 0;
  c0[i] = 0.0f;
  const int b = i >> 10, h = i & 1023;
  feats[(size_t)b * 48 * 1024 + h] = 0.0f;        // [b, t=0, h]
}

// -------------------------------------------- big GEMM: gates_x = X@Wih^T+b --
// M=24064, N=4096, K=2048. Block tile 128x128, 8 waves (4M x 2N), wave 32x64.
__global__ __launch_bounds__(256) void gemm_x_kernel(
    const unsigned short* __restrict__ X,      // [24064, 2048] bf16
    const unsigned short* __restrict__ Wih,    // [4096, 2048] bf16 (row = output n)
    const float* __restrict__ b_ih,
    const float* __restrict__ b_hh,
    float* __restrict__ gates_x)               // [24064, 4096] f32
{
  __shared__ __align__(16) unsigned int As[128 * 16];  // 128 x 32 bf16
  __shared__ __align__(16) unsigned int Bs[128 * 16];  // 128(n) x 32(k) bf16

  const int tid  = threadIdx.x;
  const int lane = tid & 31;
  const int wave = tid >> 5;
  const int half = lane >> 4;
  const int lr   = lane & 15;
  const int waveM = wave & 3;       // 32 rows each
  const int waveN = wave >> 2;      // 64 cols each
  const int m0 = blockIdx.x * 128;
  const int n0 = blockIdx.y * 128;

  const int ldrow = tid >> 1;          // 0..127
  const int ldq   = (tid & 1) * 2;     // uint4 slot base (row = 4 x uint4)

  FragC acc[2][4];
#pragma unroll
  for (int mf = 0; mf < 2; ++mf)
#pragma unroll
    for (int nf = 0; nf < 4; ++nf)
#pragma unroll
      for (int j = 0; j < 8; ++j) acc[mf][nf].f[j] = 0.0f;

  for (int k0 = 0; k0 < 2048; k0 += 32) {
    const uint4* xg = (const uint4*)(X   + (size_t)(m0 + ldrow) * 2048 + k0);
    const uint4* wg = (const uint4*)(Wih + (size_t)(n0 + ldrow) * 2048 + k0);
    const uint4 a0 = xg[ldq], a1 = xg[ldq + 1];
    const uint4 w0 = wg[ldq], w1 = wg[ldq + 1];
    __syncthreads();
    ((uint4*)(As + ldrow * 16))[ldq]     = a0;
    ((uint4*)(As + ldrow * 16))[ldq + 1] = a1;
    ((uint4*)(Bs + ldrow * 16))[ldq]     = w0;
    ((uint4*)(Bs + ldrow * 16))[ldq + 1] = w1;
    __syncthreads();

    FragAB af[2];
#pragma unroll
    for (int mf = 0; mf < 2; ++mf) {
      const unsigned int* row = As + (waveM * 32 + mf * 16 + lr) * 16;
      *(uint4*)&af[mf].u[0] = *(const uint4*)(row + 4 * half);      // K: 8h..8h+7
      *(uint4*)&af[mf].u[4] = *(const uint4*)(row + 8 + 4 * half);  // K: 16+8h..
    }
#pragma unroll
    for (int nf = 0; nf < 4; ++nf) {
      FragAB bfrag;
      const unsigned int* row = Bs + (waveN * 64 + nf * 16 + lr) * 16;
      *(uint4*)&bfrag.u[0] = *(const uint4*)(row + 4 * half);       // K: 8h..8h+7
      *(uint4*)&bfrag.u[4] = *(const uint4*)(row + 8 + 4 * half);   // K: 16+8h..
#pragma unroll
      for (int mf = 0; mf < 2; ++mf) {
        acc[mf][nf].v = __builtin_amdgcn_wmma_f32_16x16x32_bf16(
            false, af[mf].v, false, bfrag.v, (short)0, acc[mf][nf].v, false, false);
      }
    }
  }

#pragma unroll
  for (int nf = 0; nf < 4; ++nf) {
    const int n = n0 + waveN * 64 + nf * 16 + lr;
    const float bias = b_ih[n] + b_hh[n];
#pragma unroll
    for (int mf = 0; mf < 2; ++mf) {
      const int mbase = m0 + waveM * 32 + mf * 16 + 8 * half;
#pragma unroll
      for (int j = 0; j < 8; ++j)
        gates_x[(size_t)(mbase + j) * 4096 + n] = acc[mf][nf].f[j] + bias;
    }
  }
}

// ------------------------------------------------ fused recurrence step ----
// gates = gates_x[t] + h_{t-1}@Whh^T ; cell update ; write h_t.
// Block tile: 64 batch x 64 hidden, all 4 gate groups. 8 waves = 4(m) x 2(n).
// W_hh tiles staged by the Tensor Data Mover into double-buffered LDS; wave 0
// issues 4 descriptors (one per gate group) for K-slice k+1 before computing
// slice k, then s_wait_tensorcnt 0 + workgroup barrier publishes the buffer.
__global__ __launch_bounds__(256) void lstm_step_kernel(
    const unsigned short* __restrict__ Hprev,  // [512,1024] bf16
    const unsigned short* __restrict__ Whh,    // [4096,1024] bf16
    const float* __restrict__ gates_x,         // [47,512,4096] f32
    float* __restrict__ Cst,                   // [512,1024] f32
    unsigned short* __restrict__ Hnext,        // [512,1024] bf16
    float* __restrict__ feats,                 // [512,48,1024] f32 (d_out)
    int t)
{
  __shared__ __align__(16) unsigned int As[2][64 * 16];    // 2 x (64 x 32 bf16)
  __shared__ __align__(16) unsigned int Bs[2][256 * 16];   // 2 x (4 grp x 64 x 32 bf16)

  const int tid  = threadIdx.x;
  const int lane = tid & 31;
  const int wave = tid >> 5;
  const int half = lane >> 4;
  const int lr   = lane & 15;
  const int waveM = wave & 3;
  const int waveN = wave >> 2;
  const int b0  = blockIdx.x * 64;
  const int hb0 = blockIdx.y * 64;

  FragC acc[4][2];   // [gate group i,f,g,o][n-frag]
#pragma unroll
  for (int g = 0; g < 4; ++g)
#pragma unroll
    for (int nf = 0; nf < 2; ++nf)
#pragma unroll
      for (int j = 0; j < 8; ++j) acc[g][nf].f[j] = 0.0f;

  const int arow = tid >> 2, aq = tid & 3;                 // A loader mapping

  // ---- prologue: stage K-slice 0 into buffer 0 ----
  if (wave == 0) {
#pragma unroll
    for (int g = 0; g < 4; ++g) {
      tdm_load_2d_bf16(
          (unsigned int)(unsigned long long)(const void*)&Bs[0][g * 64 * 16],
          Whh + ((size_t)g * 1024 + hb0) * 1024, /*rows*/ 64, /*cols*/ 32,
          /*tensor_rows*/ 4096, /*tensor_cols*/ 1024, /*stride*/ 1024);
    }
  }
  {
    const uint4 av = ((const uint4*)(Hprev + (size_t)(b0 + arow) * 1024))[aq];
    ((uint4*)(As[0] + arow * 16))[aq] = av;
  }
  if (wave == 0) __builtin_amdgcn_s_wait_tensorcnt(0);
  __syncthreads();

  for (int kk = 0; kk < 32; ++kk) {
    const int cur = kk & 1;
    if (kk < 31) {
      const int nxt = cur ^ 1;
      const int k0n = (kk + 1) * 32;
      if (wave == 0) {
#pragma unroll
        for (int g = 0; g < 4; ++g) {
          tdm_load_2d_bf16(
              (unsigned int)(unsigned long long)(const void*)&Bs[nxt][g * 64 * 16],
              Whh + ((size_t)g * 1024 + hb0) * 1024 + k0n, 64, 32, 4096, 1024, 1024);
        }
      }
      const uint4 av = ((const uint4*)(Hprev + (size_t)(b0 + arow) * 1024 + k0n))[aq];
      ((uint4*)(As[nxt] + arow * 16))[aq] = av;
    }

    FragAB af;
    const unsigned int* arowp = As[cur] + (waveM * 16 + lr) * 16;
    *(uint4*)&af.u[0] = *(const uint4*)(arowp + 4 * half);
    *(uint4*)&af.u[4] = *(const uint4*)(arowp + 8 + 4 * half);
#pragma unroll
    for (int g = 0; g < 4; ++g) {
#pragma unroll
      for (int nf = 0; nf < 2; ++nf) {
        FragAB bfrag;
        const unsigned int* row = Bs[cur] + (g * 64 + waveN * 32 + nf * 16 + lr) * 16;
        *(uint4*)&bfrag.u[0] = *(const uint4*)(row + 4 * half);
        *(uint4*)&bfrag.u[4] = *(const uint4*)(row + 8 + 4 * half);
        acc[g][nf].v = __builtin_amdgcn_wmma_f32_16x16x32_bf16(
            false, af.v, false, bfrag.v, (short)0, acc[g][nf].v, false, false);
      }
    }

    if (wave == 0) __builtin_amdgcn_s_wait_tensorcnt(0);
    __syncthreads();   // publishes buf[nxt]; also closes reads of buf[cur]
  }

#pragma unroll
  for (int nf = 0; nf < 2; ++nf) {
    const int nH = hb0 + waveN * 32 + nf * 16 + lr;
#pragma unroll
    for (int j = 0; j < 8; ++j) {
      const int mb = b0 + waveM * 16 + 8 * half + j;
      const size_t gx = ((size_t)t * 512 + mb) * 4096;
      const float gi = acc[0][nf].f[j] + gates_x[gx + nH];
      const float gf = acc[1][nf].f[j] + gates_x[gx + 1024 + nH];
      const float gg = acc[2][nf].f[j] + gates_x[gx + 2048 + nH];
      const float go = acc[3][nf].f[j] + gates_x[gx + 3072 + nH];
      const float i_ = 1.0f / (1.0f + __expf(-gi));
      const float f_ = 1.0f / (1.0f + __expf(-gf));
      const float o_ = 1.0f / (1.0f + __expf(-go));
      const float g_ = tanhf(gg);
      const size_t ci = (size_t)mb * 1024 + nH;
      const float c_ = f_ * Cst[ci] + i_ * g_;
      Cst[ci] = c_;
      const float h_ = o_ * tanhf(c_);
      Hnext[ci] = f32_to_bf16_rne(h_);
      feats[((size_t)mb * 48 + (t + 1)) * 1024 + nH] = h_;
    }
  }
}

// -------------------------------------------------------------------- host --
extern "C" void kernel_launch(void* const* d_in, const int* in_sizes, int n_in,
                              void* d_out, int out_size, void* d_ws, size_t ws_size,
                              hipStream_t stream) {
  (void)in_sizes; (void)n_in; (void)out_size; (void)ws_size;
  const float* dh    = (const float*)d_in[0];   // [48,1,512,1024]
  const int*   caps  = (const int*)d_in[2];     // [48,512]
  const float* W_ih  = (const float*)d_in[3];   // [4096,2048]
  const float* W_hh  = (const float*)d_in[4];   // [4096,1024]
  const float* b_ih  = (const float*)d_in[5];   // [4096]
  const float* b_hh  = (const float*)d_in[6];   // [4096]
  float* feats = (float*)d_out;                 // [512,48,1024]

  char* ws = (char*)d_ws;
  size_t off = 0;
  auto carve = [&](size_t bytes) {
    char* p = ws + off;
    off = (off + bytes + 255) & ~(size_t)255;
    return p;
  };
  float*          gates_x = (float*)carve((size_t)47 * 512 * 4096 * 4);
  unsigned short* Xbf     = (unsigned short*)carve((size_t)24064 * 2048 * 2);
  unsigned short* WihBf   = (unsigned short*)carve((size_t)4096 * 2048 * 2);
  unsigned short* WhhBf   = (unsigned short*)carve((size_t)4096 * 1024 * 2);
  unsigned short* pooled  = (unsigned short*)carve((size_t)512 * 1024 * 2);
  unsigned short* hA      = (unsigned short*)carve((size_t)512 * 1024 * 2);
  unsigned short* hB      = (unsigned short*)carve((size_t)512 * 1024 * 2);
  float*          cSt     = (float*)carve((size_t)512 * 1024 * 4);

  cvt_bf16_kernel<<<(4096 * 2048) / 256, 256, 0, stream>>>(W_ih, WihBf, 4096 * 2048);
  cvt_bf16_kernel<<<(4096 * 1024) / 256, 256, 0, stream>>>(W_hh, WhhBf, 4096 * 1024);
  pool_kernel<<<512, 256, 0, stream>>>(dh, caps, pooled);
  xsprep_kernel<<<24064, 256, 0, stream>>>(dh, pooled, Xbf);
  init_kernel<<<(512 * 1024) / 256, 256, 0, stream>>>(hA, cSt, feats);
  gemm_x_kernel<<<dim3(188, 32), 256, 0, stream>>>(Xbf, WihBf, b_ih, b_hh, gates_x);
  for (int t = 0; t < 47; ++t) {
    const unsigned short* hin = (t & 1) ? hB : hA;
    unsigned short*       hout = (t & 1) ? hA : hB;
    lstm_step_kernel<<<dim3(8, 16), 256, 0, stream>>>(hin, WhhBf, gates_x, cSt, hout,
                                                      feats, t);
  }
}
